// kopt_Decoder_38603166056658
// MI455X (gfx1250) — matrix operand order repaired and checked by reading, not compiled, not run on hardware
//
#include <hip/hip_runtime.h>
#include <hip/hip_bf16.h>
#include <math.h>

#define BS 256
#define GS 1024
#define D  128
#define KMAX 4
#define NEGV (-1e30f)
#define VRANGEF 6.0f

typedef __attribute__((ext_vector_type(16))) _Float16 v16h;
typedef __attribute__((ext_vector_type(8)))  float    v8f;

__device__ __forceinline__ float sigmoidf_(float x){ return 1.0f/(1.0f+expf(-x)); }

// ---------------------------------------------------------------- setup kernels

// Convert W_K1..W_K4 (f32 row-major [128][128]) to f16, packed [4][128][128].
__global__ void k_convW(const float* __restrict__ w0, const float* __restrict__ w1,
                        const float* __restrict__ w2, const float* __restrict__ w3,
                        _Float16* __restrict__ out){
  int idx = blockIdx.x*blockDim.x + threadIdx.x;   // 65536 total
  int m = idx >> 14, r = idx & 16383;
  const float* s = (m==0)?w0:(m==1)?w1:(m==2)?w2:w3;
  out[idx] = (_Float16)s[r];
}

// V1,V2 from the meta MLP.
__global__ void __launch_bounds__(256) k_meta(const float* __restrict__ context2,
    const float* __restrict__ mW1, const float* __restrict__ mb1,
    const float* __restrict__ mW2, const float* __restrict__ mb2,
    float* __restrict__ V1, float* __restrict__ V2){
  int b = blockIdx.x, tid = threadIdx.x;
  __shared__ float sc[9], sh8[8];
  if (tid < 9) sc[tid] = context2[b*9 + tid];
  __syncthreads();
  if (tid < 8){
    float a = mb1[tid];
    for (int k=0;k<9;++k) a += sc[k]*mW1[tid*9+k];
    sh8[tid] = fmaxf(a, 0.0f);
  }
  __syncthreads();
  float a = mb2[tid];
  for (int k=0;k<8;++k) a += sh8[k]*mW2[tid*8+k];
  if (tid < D) V1[b*D + tid] = a; else V2[b*D + (tid-D)] = a;
}

__global__ void __launch_bounds__(128) k_hmean(const float* __restrict__ h,
                                               float* __restrict__ hmean){
  int b = blockIdx.x, d = threadIdx.x;
  const float* p = h + (size_t)b*GS*D + d;
  float s = 0.0f;
  for (int g=0; g<GS; ++g) s += p[(size_t)g*D];
  hmean[b*D + d] = s * (1.0f/(float)GS);
}

__global__ void __launch_bounds__(128) k_qinit(const float* __restrict__ hmean,
    const float* __restrict__ W, const float* __restrict__ bias,
    float* __restrict__ q1, float* __restrict__ q2){
  int b = blockIdx.x, n = threadIdx.x;
  __shared__ float sm[D];
  sm[n] = hmean[b*D + n];
  __syncthreads();
  float a = bias[n];
  for (int k=0;k<D;++k) a += sm[k]*W[n*D+k];
  q1[b*D+n] = a; q2[b*D+n] = a;
}

__global__ void __launch_bounds__(128) k_init(const float* __restrict__ init_query,
    float* __restrict__ inq1, float* __restrict__ inq2,
    int* __restrict__ stopped_a, int* __restrict__ nola_a, float* __restrict__ ll_a,
    int* __restrict__ aidx, int* __restrict__ kal, int* __restrict__ kar){
  int b = blockIdx.x, tid = threadIdx.x;
  float v = init_query[tid];
  inq1[b*D+tid] = v; inq2[b*D+tid] = v;
  if (tid == 0){ stopped_a[b] = 1; nola_a[b] = -1; ll_a[b] = 0.0f; }
  if (tid < KMAX){ aidx[b*KMAX+tid] = 0; kar[b*KMAX+tid] = 0; }
  if (tid < KMAX+1) kal[b*(KMAX+1)+tid] = 0;
}

__global__ void k_init_mask(int* __restrict__ mask){
  mask[blockIdx.x*256 + threadIdx.x] = 0;   // 1024 x 256 = BS*GS
}

// ---------------------------------------------------------------- per-step kernels

// One block per (b, rnn); 384 threads (one per gate output).
__global__ void __launch_bounds__(384) k_gru(
    const float* __restrict__ Wih1, const float* __restrict__ Whh1,
    const float* __restrict__ bih1, const float* __restrict__ bhh1,
    const float* __restrict__ Wih2, const float* __restrict__ Whh2,
    const float* __restrict__ bih2, const float* __restrict__ bhh2,
    float* __restrict__ q1, float* __restrict__ q2,
    const float* __restrict__ inq1, const float* __restrict__ inq2){
  int b = blockIdx.x, rsel = blockIdx.y, tid = threadIdx.x;
  const float* Wih = rsel ? Wih2 : Wih1;
  const float* Whh = rsel ? Whh2 : Whh1;
  const float* bih = rsel ? bih2 : bih1;
  const float* bhh = rsel ? bhh2 : bhh1;
  float* q = rsel ? q2 : q1;
  const float* x = (rsel ? inq2 : inq1) + b*D;
  __shared__ float sx[D], sh[D], si[3*D], sg[3*D];
  if (tid < D){ sx[tid] = x[tid]; sh[tid] = q[b*D+tid]; }
  __syncthreads();
  float ai = bih[tid], ah = bhh[tid];
  for (int k=0;k<D;++k){ ai += sx[k]*Wih[tid*D+k]; ah += sh[k]*Whh[tid*D+k]; }
  si[tid] = ai; sg[tid] = ah;
  __syncthreads();
  if (tid < D){
    float rr = sigmoidf_(si[tid]       + sg[tid]);
    float z  = sigmoidf_(si[tid+D]     + sg[tid+D]);
    float nn = tanhf   (si[tid+2*D] + rr*sg[tid+2*D]);
    q[b*D+tid] = (1.0f - z)*nn + z*sh[tid];
  }
}

// qq1=q1@WQ1^T, qq2=q2@WQ2^T, qq3=q1@WQ3^T, qq4=q2@WQ4^T. grid (BS,4) x 128.
__global__ void __launch_bounds__(128) k_qproj(
    const float* __restrict__ WQ1, const float* __restrict__ WQ2,
    const float* __restrict__ WQ3, const float* __restrict__ WQ4,
    const float* __restrict__ q1, const float* __restrict__ q2,
    float* __restrict__ qq1, float* __restrict__ qq2,
    float* __restrict__ qq3, float* __restrict__ qq4){
  int b = blockIdx.x, p = blockIdx.y, tid = threadIdx.x;
  const float* W = (p==0)?WQ1:(p==1)?WQ2:(p==2)?WQ3:WQ4;
  const float* q = (p==0||p==2) ? q1 : q2;
  float* out = (p==0)?qq1:(p==1)?qq2:(p==2)?qq3:qq4;
  __shared__ float sq[D];
  sq[tid] = q[b*D+tid];
  __syncthreads();
  float acc = 0.0f;
  for (int k=0;k<D;++k) acc += sq[k]*W[tid*D+k];
  out[b*D+tid] = acc;
}

// Fused: K-projections (WMMA f16->f32) + tanh attention + dot(V) + logits.
// Block = one 16-row tile (16 consecutive g of one b); 8 waves each own a 16-col d-tile.
__global__ void __launch_bounds__(256) k_logits(
    const float* __restrict__ h, const _Float16* __restrict__ Wh,   // [4][128][128] f16
    const float* __restrict__ qq1, const float* __restrict__ qq2,
    const float* __restrict__ qq3, const float* __restrict__ qq4,
    const float* __restrict__ V1, const float* __restrict__ V2,
    float* __restrict__ logits){
  __shared__ _Float16 sA[16*D];
  __shared__ float sRes[16];
  int tid = threadIdx.x;
  int wave = tid >> 5, lane = tid & 31;
  size_t row0 = (size_t)blockIdx.x * 16;
  int b  = (int)(row0 / GS);
  int g0 = (int)(row0 % GS);
  const float* hp = h + row0 * D;
  // Stage the 16x128 f32 h-tile into LDS as f16 (coalesced, 8 elems/thread).
  #pragma unroll
  for (int e=0;e<8;++e){
    int idx = tid*8 + e;
    sA[idx] = (_Float16)hp[idx];
  }
  if (tid < 16) sRes[tid] = 0.0f;
  __syncthreads();

  v8f acc[4];
  #pragma unroll
  for (int m=0;m<4;++m)
    #pragma unroll
    for (int j=0;j<8;++j) acc[m][j] = 0.0f;

  int r    = lane & 15;
  int ncol = (wave << 4) + (lane & 15);      // output d-column of this lane
  #pragma unroll
  for (int ks=0; ks<D; ks+=32){
    // A fragment (16x32 f16): lanes 0-15 K=0..15, lanes 16-31 K=16..31.
    v16h a;
    int ka = ks + ((lane >= 16) ? 16 : 0);
    #pragma unroll
    for (int e=0;e<16;++e){
      int k = ka + e;
      a[e] = sA[r*D + k];
    }
    // B fragments (32x16 f16): B[k][n] = W[n][k]; lanes 0-15 K=0..15, 16-31 K=16..31.
    int kb = ks + ((lane >= 16) ? 16 : 0);
    v16h bf0, bf1, bf2, bf3;
    #pragma unroll
    for (int e=0;e<16;++e){
      bf0[e] = Wh[0*D*D + ncol*D + kb + e];
      bf1[e] = Wh[1*D*D + ncol*D + kb + e];
      bf2[e] = Wh[2*D*D + ncol*D + kb + e];
      bf3[e] = Wh[3*D*D + ncol*D + kb + e];
    }
    acc[0] = __builtin_amdgcn_wmma_f32_16x16x32_f16(false, a, false, bf0, (short)0, acc[0], false, false);
    acc[1] = __builtin_amdgcn_wmma_f32_16x16x32_f16(false, a, false, bf1, (short)0, acc[1], false, false);
    acc[2] = __builtin_amdgcn_wmma_f32_16x16x32_f16(false, a, false, bf2, (short)0, acc[2], false, false);
    acc[3] = __builtin_amdgcn_wmma_f32_16x16x32_f16(false, a, false, bf3, (short)0, acc[3], false, false);
  }
  // acc[m][j] = K{m+1}h[row, ncol], row = j + (lane>=16 ? 8 : 0)
  float p1 = qq1[b*D+ncol], p2 = qq2[b*D+ncol], p3 = qq3[b*D+ncol], p4 = qq4[b*D+ncol];
  float v1 = V1[b*D+ncol],  v2 = V2[b*D+ncol];
  int rb = (lane >= 16) ? 8 : 0;
  #pragma unroll
  for (int j=0;j<8;++j){
    float t1 = tanhf(acc[0][j] + p1 + acc[2][j]*p3);
    float t2 = tanhf(acc[1][j] + p2 + acc[3][j]*p4);
    atomicAdd(&sRes[rb + j], t1*v1 + t2*v2);      // ds_add_f32 reduce over d
  }
  __syncthreads();
  if (tid < 16) logits[(size_t)b*GS + g0 + tid] = tanhf(sRes[tid]) * VRANGEF;
}

// Sequential selection / masking step. One block per b, 256 threads (4 g each).
__global__ void __launch_bounds__(256) k_select(int iter,
    const float* __restrict__ logits, const int* __restrict__ rec,
    const int* __restrict__ visited_time, const int* __restrict__ last_action,
    const int* __restrict__ fixed_action, const float* __restrict__ h,
    int* __restrict__ mask, int* __restrict__ vtt,
    int* __restrict__ stopped_a, int* __restrict__ nola_a, float* __restrict__ ll_a,
    int* __restrict__ aidx, int* __restrict__ kal, int* __restrict__ kar,
    float* __restrict__ inq1, float* __restrict__ inq2){
  int b = blockIdx.x, tid = threadIdx.x;
  __shared__ float red[256];
  __shared__ float s_actlogit;
  __shared__ int   s_vt0;
  __shared__ int   s_vtta;

  int old_stopped = stopped_a[b];
  int old_nola    = nola_a[b];
  int aidx0_prev  = aidx[b*KMAX + 0];
  int act;
  if (iter == 0) act = fixed_action[b*KMAX + 0];
  else           act = old_stopped ? aidx0_prev : fixed_action[b*KMAX + iter];
  int aidx0 = (iter == 0) ? act : aidx0_prev;
  int la = last_action[b];

  float lg[4];
  for (int u=0;u<4;++u){
    int g = tid + u*256;
    float v = logits[(size_t)b*GS + g];
    if (mask[(size_t)b*GS + g]) v = NEGV;
    if (iter == 0 && g == la) v = NEGV;
    lg[u] = v;
    if (g == act) s_actlogit = v;
  }
  if (tid == 0 && iter == 0) s_vt0 = visited_time[(size_t)b*GS + act];
  __syncthreads();

  // log-softmax: max then sum-exp (block tree reductions)
  float m = fmaxf(fmaxf(lg[0],lg[1]), fmaxf(lg[2],lg[3]));
  red[tid] = m; __syncthreads();
  for (int s=128;s>0;s>>=1){ if (tid<s) red[tid]=fmaxf(red[tid],red[tid+s]); __syncthreads(); }
  float mx = red[0]; __syncthreads();
  float ssum = expf(lg[0]-mx)+expf(lg[1]-mx)+expf(lg[2]-mx)+expf(lg[3]-mx);
  red[tid] = ssum; __syncthreads();
  for (int s=128;s>0;s>>=1){ if (tid<s) red[tid]+=red[tid+s]; __syncthreads(); }
  float lse = mx + logf(red[0]);
  float loss_now = s_actlogit - lse;

  int nona = rec[(size_t)b*GS + act];
  int hit = (act == old_nola) ? 1 : 0;
  int new_stopped = (iter > 0) ? (old_stopped | hit) : hit;

  // gathers use OLD stopped / OLD nola
  int nolam = ((old_nola % GS) + GS) % GS;
  if (tid < D){
    float x1 = h[((size_t)b*GS + act)*D + tid];
    inq1[b*D+tid] = x1;
    inq2[b*D+tid] = old_stopped ? x1 : h[((size_t)b*GS + nolam)*D + tid];
  }
  if (iter == 0){
    for (int u=0;u<4;++u){
      int g = tid + u*256;
      int dv = visited_time[(size_t)b*GS + g] - s_vt0;
      vtt[(size_t)b*GS + g] = ((dv % GS) + GS) % GS;
    }
  }
  if (tid == 0){
    float add = (iter == 0) ? loss_now : (old_stopped ? 0.0f : loss_now);
    ll_a[b] += add;
    aidx[b*KMAX + iter] = act;
    int jprev = (iter - 1 + KMAX) % KMAX;
    int kv = kal[b*(KMAX+1) + iter];
    kal[b*(KMAX+1) + iter] = old_stopped ? act : kv;
    if (!old_stopped) kar[b*KMAX + jprev] = act;
    kal[b*(KMAX+1) + iter + 1] = nona;
    if (new_stopped){
      kal[b*(KMAX+1) + iter] = kal[b*(KMAX+1) + ((iter - 1 + (KMAX+1)) % (KMAX+1))];
      kar[b*KMAX + iter]     = kar[b*KMAX + jprev];
    }
    stopped_a[b] = new_stopped;
    nola_a[b]    = new_stopped ? -1 : nona;
  }
  __syncthreads();
  if (tid == 0) s_vtta = vtt[(size_t)b*GS + act];
  __syncthreads();
  int allow = (!new_stopped) && (nona == aidx0);
  for (int u=0;u<4;++u){
    int g = tid + u*256;
    int vg = vtt[(size_t)b*GS + g];
    int mk = (vg <= s_vtta) ? 1 : 0;
    if (iter == 0 && vg > GS-2) mk = 1;
    if (new_stopped && g == act) mk = 0;
    if (allow && g == aidx0) mk = 0;
    mask[(size_t)b*GS + g] = mk;
  }
}

__global__ void k_finalize(const int* __restrict__ stopped_a, const int* __restrict__ aidx,
                           int* __restrict__ kal, int* __restrict__ kar,
                           const float* __restrict__ ll_a, float* __restrict__ out){
  int b = threadIdx.x;  // 256
  if (!stopped_a[b]) kar[b*KMAX + (KMAX-1)] = kal[b*(KMAX+1) + KMAX];
  for (int j=0;j<KMAX;++j) out[b*(3*KMAX) + j]          = (float)aidx[b*KMAX + j];
  for (int j=0;j<KMAX;++j) out[b*(3*KMAX) + KMAX + j]   = (float)kal[b*(KMAX+1) + j];
  for (int j=0;j<KMAX;++j) out[b*(3*KMAX) + 2*KMAX + j] = (float)kar[b*KMAX + j];
  out[BS*(3*KMAX) + b] = ll_a[b];
}

// ---------------------------------------------------------------- host

extern "C" void kernel_launch(void* const* d_in, const int* in_sizes, int n_in,
                              void* d_out, int out_size, void* d_ws, size_t ws_size,
                              hipStream_t stream){
  (void)in_sizes; (void)n_in; (void)out_size; (void)ws_size;
  const float* h            = (const float*)d_in[0];
  const float* context2     = (const float*)d_in[1];
  const int*   rec          = (const int*)d_in[2];
  const int*   visited_time = (const int*)d_in[3];
  const int*   last_action  = (const int*)d_in[4];
  const int*   fixed_action = (const int*)d_in[5];
  // d_in[6] = k_max (compile-time KMAX)
  const float* W_K1 = (const float*)d_in[7];
  const float* W_K2 = (const float*)d_in[8];
  const float* W_K3 = (const float*)d_in[9];
  const float* W_K4 = (const float*)d_in[10];
  const float* W_Q1 = (const float*)d_in[11];
  const float* W_Q2 = (const float*)d_in[12];
  const float* W_Q3 = (const float*)d_in[13];
  const float* W_Q4 = (const float*)d_in[14];
  const float* meta_W1 = (const float*)d_in[15];
  const float* meta_b1 = (const float*)d_in[16];
  const float* meta_W2 = (const float*)d_in[17];
  const float* meta_b2 = (const float*)d_in[18];
  const float* init_hidden_W = (const float*)d_in[19];
  const float* init_hidden_b = (const float*)d_in[20];
  const float* init_query    = (const float*)d_in[21];
  const float* rnn1_Wih = (const float*)d_in[22];
  const float* rnn1_Whh = (const float*)d_in[23];
  const float* rnn1_bih = (const float*)d_in[24];
  const float* rnn1_bhh = (const float*)d_in[25];
  const float* rnn2_Wih = (const float*)d_in[26];
  const float* rnn2_Whh = (const float*)d_in[27];
  const float* rnn2_bih = (const float*)d_in[28];
  const float* rnn2_bhh = (const float*)d_in[29];

  char* wp = (char*)d_ws;
  auto take = [&](size_t bytes)->void*{
    void* p = (void*)wp; wp += (bytes + 255) & ~(size_t)255; return p;
  };
  _Float16* Whalf = (_Float16*)take((size_t)4*D*D*sizeof(_Float16));
  float* logits = (float*)take((size_t)BS*GS*sizeof(float));
  int*   vtt    = (int*)  take((size_t)BS*GS*sizeof(int));
  int*   mask   = (int*)  take((size_t)BS*GS*sizeof(int));
  float* q1     = (float*)take((size_t)BS*D*sizeof(float));
  float* q2     = (float*)take((size_t)BS*D*sizeof(float));
  float* inq1   = (float*)take((size_t)BS*D*sizeof(float));
  float* inq2   = (float*)take((size_t)BS*D*sizeof(float));
  float* qq1    = (float*)take((size_t)BS*D*sizeof(float));
  float* qq2    = (float*)take((size_t)BS*D*sizeof(float));
  float* qq3    = (float*)take((size_t)BS*D*sizeof(float));
  float* qq4    = (float*)take((size_t)BS*D*sizeof(float));
  float* V1     = (float*)take((size_t)BS*D*sizeof(float));
  float* V2     = (float*)take((size_t)BS*D*sizeof(float));
  float* hmean  = (float*)take((size_t)BS*D*sizeof(float));
  int*   stopped= (int*)  take((size_t)BS*sizeof(int));
  int*   nola   = (int*)  take((size_t)BS*sizeof(int));
  float* ll     = (float*)take((size_t)BS*sizeof(float));
  int*   aidx   = (int*)  take((size_t)BS*KMAX*sizeof(int));
  int*   kal    = (int*)  take((size_t)BS*(KMAX+1)*sizeof(int));
  int*   kar    = (int*)  take((size_t)BS*KMAX*sizeof(int));

  k_convW<<<256, 256, 0, stream>>>(W_K1, W_K2, W_K3, W_K4, Whalf);
  k_meta <<<BS, 256, 0, stream>>>(context2, meta_W1, meta_b1, meta_W2, meta_b2, V1, V2);
  k_hmean<<<BS, 128, 0, stream>>>(h, hmean);
  k_qinit<<<BS, 128, 0, stream>>>(hmean, init_hidden_W, init_hidden_b, q1, q2);
  k_init <<<BS, 128, 0, stream>>>(init_query, inq1, inq2, stopped, nola, ll, aidx, kal, kar);
  k_init_mask<<<1024, 256, 0, stream>>>(mask);

  for (int i=0; i<KMAX; ++i){
    k_gru  <<<dim3(BS,2), 384, 0, stream>>>(rnn1_Wih, rnn1_Whh, rnn1_bih, rnn1_bhh,
                                            rnn2_Wih, rnn2_Whh, rnn2_bih, rnn2_bhh,
                                            q1, q2, inq1, inq2);
    k_qproj<<<dim3(BS,4), 128, 0, stream>>>(W_Q1, W_Q2, W_Q3, W_Q4, q1, q2, qq1, qq2, qq3, qq4);
    k_logits<<<(BS*GS)/16, 256, 0, stream>>>(h, Whalf, qq1, qq2, qq3, qq4, V1, V2, logits);
    k_select<<<BS, 256, 0, stream>>>(i, logits, rec, visited_time, last_action, fixed_action, h,
                                     mask, vtt, stopped, nola, ll, aidx, kal, kar, inq1, inq2);
  }
  k_finalize<<<1, 256, 0, stream>>>(stopped, aidx, kal, kar, ll, (float*)d_out);
}
